// Contract18Module_29729763623443
// MI455X (gfx1250) — matrix operand order repaired and checked by reading, not compiled, not run on hardware
//
#include <hip/hip_runtime.h>
#include <hip/hip_bf16.h>
#include <stddef.h>

#define N   256
#define C   16
#define NCH 288   // 18 * C

typedef __attribute__((ext_vector_type(16))) _Float16 v16h;
typedef __attribute__((ext_vector_type(8)))  _Float16 v8h;
typedef __attribute__((ext_vector_type(2)))  _Float16 v2h;
typedef __attribute__((ext_vector_type(8)))  float    v8f;

static __device__ __forceinline__ v16h pack16(v8h lo, v8h hi) {
  v16h r;
#pragma unroll
  for (int s = 0; s < 8; ++s) { r[s] = lo[s]; r[s + 8] = hi[s]; }
  return r;
}

// ---------- zero channels 0..95 (atomic-accumulated c1..c6 blocks) ----------
__global__ void c18_zero_kernel(float* __restrict__ out) {
  size_t idx = (size_t)blockIdx.x * blockDim.x + threadIdx.x;
  const size_t total = (size_t)N * N * 24;        // 24 float4 per (a,b)
  if (idx >= total) return;
  size_t pair = idx / 24, q = idx % 24;
  *(float4*)(out + pair * NCH + q * 4) = make_float4(0.f, 0.f, 0.f, 0.f);
}

// ---------- diagonal extractions c7..c12 (channels 96..191) ----------------
__global__ void c18_diag_kernel(const float* __restrict__ T, float* __restrict__ out) {
  int idx = blockIdx.x * blockDim.x + threadIdx.x;
  if (idx >= N * N) return;
  int a = idx >> 8, b = idx & 255;
  const size_t srcs[6] = {
    (((size_t)a * N + a) * N + b) * C,   // c7  T[a,a,b]
    (((size_t)b * N + b) * N + a) * C,   // c8  T[b,b,a]
    (((size_t)a * N + b) * N + a) * C,   // c9  T[a,b,a]
    (((size_t)b * N + a) * N + b) * C,   // c10 T[b,a,b]
    (((size_t)a * N + b) * N + b) * C,   // c11 T[a,b,b]
    (((size_t)b * N + a) * N + a) * C    // c12 T[b,a,a]
  };
  float* dst = out + (size_t)idx * NCH + 6 * C;
#pragma unroll
  for (int g = 0; g < 6; ++g) {
    const float4* s = (const float4*)(T + srcs[g]);
    float4* d = (float4*)(dst + g * C);
#pragma unroll
    for (int q = 0; q < 4; ++q) d[q] = s[q];
  }
}

// ---------- main single-pass over T: c1,c2 (f32 regs) + c3..c6 (WMMA) ------
// WMMA K-dimension mapping (our choice): K = 2*j + csel  (channel-interleaved)
// so a thread's packed (c0,c1) pair is one b32 LDS store and operand rows are
// contiguous b128 runs. LDS rows padded to 40 halves (80 B, 16 B-aligned) for
// conflict-free stores.
__global__ __launch_bounds__(256)
void c18_main_kernel(const float* __restrict__ T,
                     const float* __restrict__ A,
                     float* __restrict__ out) {
  __shared__ float sAtile[16][16];                        // A[gi+i, gj+j]
  __shared__ float sAi[16][64];                           // A[gi+i, gk0+kl]
  __shared__ float sAj[16][64];                           // A[gj+j, gk0+kl]
  __shared__ __align__(16) _Float16 sX [2][8][16][40];    // [buf][cpair][i][2*j+csel]
  __shared__ __align__(16) _Float16 sXT[2][8][16][40];    // [buf][cpair][j][2*i+csel]

  const int tid  = threadIdx.x;
  const int lane = tid & 31;
  const int wv   = tid >> 5;            // wave -> channel pair
  const int it   = tid >> 4;            // i_loc
  const int jt   = tid & 15;            // j_loc
  const int gi   = blockIdx.x * 16;
  const int gj   = blockIdx.y * 16;
  const int gk0  = blockIdx.z * 64;

  sAtile[it][jt] = A[(size_t)(gi + it) * N + (gj + jt)];
#pragma unroll
  for (int q = 0; q < 4; ++q) {
    int x = tid * 4 + q;                // 0..1023
    int r = x >> 6, kl = x & 63;
    sAi[r][kl] = A[(size_t)(gi + r) * N + (gk0 + kl)];
    sAj[r][kl] = A[(size_t)(gj + r) * N + (gk0 + kl)];
  }
  __syncthreads();

  const int  m      = lane & 15;        // WMMA row (M) and D column (n)
  const bool hiHalf = lane >= 16;
  const bool evenLn = (lane & 1) == 0;
  const int  jwB    = lane >> 1;

  // B selector: column n sums rows with csel == n (even/odd K).
  // B layout: lane%16 = column n, half-slot s = K (table order), csel = s&1.
  v16h bsel = {};
#pragma unroll
  for (int s = 0; s < 16; ++s) bsel[s] = (_Float16)(((s & 1) == m) ? 1.f : 0.f);

  // Adjacency-tile packed A-operands (Hadamard factors), loop invariant.
  // A-operand half-slot s maps to K = (s<8 ? s : s+8) + (hiHalf ? 8 : 0),
  // and K -> j = K>>1 (csel irrelevant: same tile weight for both channels).
  v16h atOp, atTOp;
#pragma unroll
  for (int s = 0; s < 16; ++s) {
    int K  = ((s < 8) ? s : s + 8) + (hiHalf ? 8 : 0);
    int jj = K >> 1;
    atOp[s]  = (_Float16)sAtile[m][jj];   // weight A[i=m, j]  (row reduce)
    atTOp[s] = (_Float16)sAtile[jj][m];   // weight A[i, j=m]  (col reduce)
  }

  const int c0 = wv * 2;

  float acc1[16], acc2[16];
#pragma unroll
  for (int c = 0; c < 16; ++c) { acc1[c] = 0.f; acc2[c] = 0.f; }

  const float* tp = T + ((((size_t)(gi + it)) * N + (gj + jt)) * N + gk0) * C;

  for (int kl = 0; kl < 64; ++kl) {
    const int buf = kl & 1;
    const float* p = tp + (size_t)kl * C;
    __builtin_prefetch(p + 8 * C, 0, 0);

    float xv[16];
#pragma unroll
    for (int q = 0; q < 4; ++q) {
      float4 v = ((const float4*)p)[q];
      xv[q*4+0] = v.x; xv[q*4+1] = v.y; xv[q*4+2] = v.z; xv[q*4+3] = v.w;
    }
    const float wi = sAi[it][kl];   // A[gi+it, gk]
    const float wj = sAj[jt][kl];   // A[gj+jt, gk]
#pragma unroll
    for (int c = 0; c < 16; ++c) {
      acc1[c] = fmaf(xv[c], wi, acc1[c]);          // c1[i,j,c]
      acc2[c] = fmaf(xv[c], wj, acc2[c]);          // c2[j,i,c]
    }
    // packed f16 channel-pair stores: 16 conflict-free ds_store_b32 total
#pragma unroll
    for (int cp = 0; cp < 8; ++cp) {
      v2h h;
      h[0] = (_Float16)xv[2 * cp];
      h[1] = (_Float16)xv[2 * cp + 1];
      *(v2h*)&sX [buf][cp][it][2 * jt] = h;
      *(v2h*)&sXT[buf][cp][jt][2 * it] = h;
    }
    __syncthreads();   // double-buffered: one barrier per iteration is safe

    // Build f16 A-operands: two 16B-aligned b128 LDS reads per operand.
    // lanes 0-15 need j in {0..3}u{8..11}, lanes 16-31 j in {4..7}u{12..15}.
    const int h0 = hiHalf ? 8 : 0;       // half offset of first run (2*j0)
    v8h xlo = *(const v8h*)&sX [buf][wv][m][h0];
    v8h xhi = *(const v8h*)&sX [buf][wv][m][h0 + 16];
    v8h tlo = *(const v8h*)&sXT[buf][wv][m][h0];
    v8h thi = *(const v8h*)&sXT[buf][wv][m][h0 + 16];
    v16h ax  = pack16(xlo, xhi);       // X[i=m, (j,csel)(K), c0+csel]
    v16h axt = pack16(tlo, thi);       // X[(i,csel)(K), j=m, c0+csel]
    v16h ah  = ax  * atOp;             // Hadamard with A-tile (v_pk_mul_f16)
    v16h aht = axt * atTOp;

    v16h b4 = {}, b6 = {};
    {
#pragma unroll
      for (int s = 0; s < 16; ++s) {
        int K = ((s < 8) ? s : s + 8) + (hiHalf ? 8 : 0);
        bool on = (s & 1) == m;
        b4[s] = on ? (_Float16)sAj[K >> 1][kl] : (_Float16)0.f;  // A[gk, gj+j] (A symmetric)
        b6[s] = on ? (_Float16)sAi[K >> 1][kl] : (_Float16)0.f;  // A[gk, gi+i]
      }
    }

    v8f z = {};
    // col0 = reduction for channel c0, col1 = channel c0+1.
    v8f d3 = __builtin_amdgcn_wmma_f32_16x16x32_f16(false, ah,  false, bsel, (short)0, z, false, false); // p3[i,c]
    v8f d4 = __builtin_amdgcn_wmma_f32_16x16x32_f16(false, ax,  false, b4,   (short)0, z, false, false); // p4[i,c]
    v8f d5 = __builtin_amdgcn_wmma_f32_16x16x32_f16(false, aht, false, bsel, (short)0, z, false, false); // p5[j,c]
    v8f d6 = __builtin_amdgcn_wmma_f32_16x16x32_f16(false, axt, false, b6,   (short)0, z, false, false); // p6[j,c]

    // D layout: lane = column n, VGPR r = row M (+8 for lanes 16..31).
    if (m < 2) {
      const int cc = c0 + m;
      const int mb = hiHalf ? 8 : 0;
      const int kg = gk0 + kl;
#pragma unroll
      for (int r = 0; r < 8; ++r) {
        int M = mb + r;
        atomicAdd(out + ((size_t)(gi + M) * N + kg) * NCH + 2 * C + cc, d3[r]); // c3[i,k]
        atomicAdd(out + ((size_t)kg * N + (gi + M)) * NCH + 3 * C + cc, d4[r]); // c4[k,i]
        atomicAdd(out + ((size_t)(gj + M) * N + kg) * NCH + 4 * C + cc, d5[r]); // c5[j,k]
        atomicAdd(out + ((size_t)kg * N + (gj + M)) * NCH + 5 * C + cc, d6[r]); // c6[k,j]
      }
    }
  }

  // flush k-split partials of c1/c2
#pragma unroll
  for (int c = 0; c < 16; ++c) {
    atomicAdd(out + ((size_t)(gi + it) * N + (gj + jt)) * NCH + c,     acc1[c]);
    atomicAdd(out + ((size_t)(gj + jt) * N + (gi + it)) * NCH + C + c, acc2[c]);
  }
}

// ---------- v1/v2/v3 from finished c1/c2/c6 blocks -------------------------
__global__ __launch_bounds__(256)
void c18_vreduce_kernel(const float* __restrict__ A, const float* __restrict__ out,
                        float* __restrict__ vbuf) {
  __shared__ float red[8][48];
  const int a = blockIdx.x;
  const int b = threadIdx.x;
  const float w = A[(size_t)a * N + b];
  const float* o = out + ((size_t)a * N + b) * NCH;
  float acc[48];
#pragma unroll
  for (int c = 0; c < 16; ++c) {
    acc[c]      = w * o[c];           // v1 from c1
    acc[16 + c] = w * o[C + c];       // v2 from c2
    acc[32 + c] = w * o[5 * C + c];   // v3 from c6
  }
#pragma unroll
  for (int off = 16; off > 0; off >>= 1)
#pragma unroll
    for (int c = 0; c < 48; ++c)
      acc[c] += __shfl_down(acc[c], off, 32);
  if ((threadIdx.x & 31) == 0) {
    int wv = threadIdx.x >> 5;
#pragma unroll
    for (int c = 0; c < 48; ++c) red[wv][c] = acc[c];
  }
  __syncthreads();
  if (threadIdx.x < 48) {
    float s = 0.f;
#pragma unroll
    for (int wv = 0; wv < 8; ++wv) s += red[wv][threadIdx.x];
    vbuf[(size_t)a * 48 + threadIdx.x] = s;
  }
}

// ---------- broadcast c13..c18 (channels 192..287) -------------------------
__global__ void c18_broadcast_kernel(const float* __restrict__ vbuf, float* __restrict__ out) {
  int idx = blockIdx.x * blockDim.x + threadIdx.x;
  if (idx >= N * N) return;
  int a = idx >> 8, b = idx & 255;
  const float* va = vbuf + (size_t)a * 48;
  const float* vb = vbuf + (size_t)b * 48;
  float* d = out + (size_t)idx * NCH + 12 * C;
#pragma unroll
  for (int c = 0; c < 16; ++c) {
    d[c]         = va[c];        // c13 = v1[a]
    d[C + c]     = vb[c];        // c14 = v1[b]
    d[2 * C + c] = va[16 + c];   // c15 = v2[a]
    d[3 * C + c] = vb[16 + c];   // c16 = v2[b]
    d[4 * C + c] = va[32 + c];   // c17 = v3[a]
    d[5 * C + c] = vb[32 + c];   // c18 = v3[b]
  }
}

extern "C" void kernel_launch(void* const* d_in, const int* in_sizes, int n_in,
                              void* d_out, int out_size, void* d_ws, size_t ws_size,
                              hipStream_t stream) {
  const float* T = (const float*)d_in[0];   // (256,256,256,16) f32
  const float* A = (const float*)d_in[1];   // (256,256) f32, symmetric 0/1
  float* out  = (float*)d_out;              // (256,256,288) f32
  float* vbuf = (float*)d_ws;               // 256*48 floats

  {
    size_t total = (size_t)N * N * 24;
    c18_zero_kernel<<<(int)((total + 255) / 256), 256, 0, stream>>>(out);
  }
  c18_diag_kernel<<<(N * N) / 256, 256, 0, stream>>>(T, out);
  {
    dim3 grid(16, 16, 4);
    c18_main_kernel<<<grid, 256, 0, stream>>>(T, A, out);
  }
  c18_vreduce_kernel<<<N, 256, 0, stream>>>(A, out, vbuf);
  c18_broadcast_kernel<<<(N * N) / 256, 256, 0, stream>>>(vbuf, out);
}
